// MultilayerGRU_54597624266954
// MI455X (gfx1250) — compile-verified, matrix operand order checked by reading, not run on hardware
//
#include <hip/hip_runtime.h>
#include <hip/hip_bf16.h>
#include <math.h>

#define B_ 32
#define S_ 512
#define I_ 128
#define H_ 1024
#define O_ 128
#define L_ 2
#define YSIZE (B_ * S_ * O_)

typedef __attribute__((ext_vector_type(16))) _Float16 v16h;
typedef __attribute__((ext_vector_type(8)))  float    v8f;
typedef __attribute__((ext_vector_type(4)))  float    f32x4;

// Per-lane WMMA 16-bit fragment: two contiguous 16B runs from a row-major
// matrix. Caller passes p = base + row*ld + ktile*32 + (lane>>4)*8.
static __device__ __forceinline__ v16h ldfrag(const _Float16* p) {
  union { v16h h; f32x4 f[2]; } u;
  u.f[0] = *reinterpret_cast<const f32x4*>(p);       // K offsets +0..7
  u.f[1] = *reinterpret_cast<const f32x4*>(p + 16);  // K offsets +16..23
  return u.h;
}

static __device__ __forceinline__ v8f wmma_f16(v16h a, v16h b, v8f c) {
  return __builtin_amdgcn_wmma_f32_16x16x32_f16(false, a, false, b,
                                                (short)0, c, false, false);
}

// ---------------------------------------------------------------- cvt f32->f16
__global__ void cvt_f32_f16_kernel(const float* __restrict__ s,
                                   _Float16* __restrict__ d, int n) {
  int i = blockIdx.x * blockDim.x + threadIdx.x;
  if (i < n) d[i] = (_Float16)s[i];
}

// ---------------------------------------------------------------- big GEMM
// D[m,n] = sum_k A[m,k] * W[n,k] (+ bias[n]).  A:(M,K) f16 row-major,
// W:(N,K) f16 row-major. Register-blocked: each wave computes a 64x32
// super-tile (4 M-tiles x 2 N-tiles) -> 8 WMMAs per 6 fragment loads.
#define GM 4
#define GN 2
__global__ void wmma_gemm_kernel(const _Float16* __restrict__ A,
                                 const _Float16* __restrict__ Bm,
                                 const float* __restrict__ bias,
                                 _Float16* __restrict__ outh,
                                 float* __restrict__ outf,
                                 int M, int N, int K) {
  const int lane = threadIdx.x & 31;
  const int wave = blockIdx.x * (blockDim.x >> 5) + (threadIdx.x >> 5);
  const int nsup = N >> 5;                // super-tiles along N (32 wide)
  const int ms   = wave / nsup;           // super-tile row (64 wide)
  const int ns   = wave - ms * nsup;
  if (ms * 64 >= M) return;               // whole-wave uniform

  const int koff = (lane >> 4) << 3;
  const _Float16* ap = A  + (size_t)(ms * 64 + (lane & 15)) * K + koff;
  const _Float16* bp = Bm + (size_t)(ns * 32 + (lane & 15)) * K + koff;

  v8f acc[GM][GN];
#pragma unroll
  for (int i = 0; i < GM; ++i)
#pragma unroll
    for (int j = 0; j < GN; ++j) acc[i][j] = (v8f){};

  for (int k = 0; k < K; k += 32) {
    __builtin_prefetch(bp + k + 128, 0, 1);   // global_prefetch_b8
    v16h bfr[GN];
#pragma unroll
    for (int j = 0; j < GN; ++j) bfr[j] = ldfrag(bp + (size_t)j * 16 * K + k);
#pragma unroll
    for (int i = 0; i < GM; ++i) {
      v16h a = ldfrag(ap + (size_t)i * 16 * K + k);
#pragma unroll
      for (int j = 0; j < GN; ++j) acc[i][j] = wmma_f16(a, bfr[j], acc[i][j]);
    }
  }

  const int half = lane >> 4;
#pragma unroll
  for (int i = 0; i < GM; ++i) {
#pragma unroll
    for (int j = 0; j < GN; ++j) {
      const int n = ns * 32 + j * 16 + (lane & 15);
      const float bv = bias ? bias[n] : 0.0f;
#pragma unroll
      for (int r = 0; r < 8; ++r) {
        const int m = ms * 64 + i * 16 + r + 8 * half;
        const float v = acc[i][j][r] + bv;
        const size_t idx = (size_t)m * N + n;
        if (outf) outf[idx] = v;
        if (outh) outh[idx] = (_Float16)v;
      }
    }
  }
}

// ---------------------------------------------------------------- grid barrier
__global__ void init_sync_kernel(int* p) { p[0] = 0; p[1] = 0; }

static __device__ __forceinline__ void grid_sync(int* cnt, int* gen, int nblocks) {
  __threadfence();
  __syncthreads();
  if (threadIdx.x == 0) {
    int g = __hip_atomic_load(gen, __ATOMIC_RELAXED, __HIP_MEMORY_SCOPE_AGENT);
    int prev = __hip_atomic_fetch_add(cnt, 1, __ATOMIC_ACQ_REL, __HIP_MEMORY_SCOPE_AGENT);
    if (prev == nblocks - 1) {
      __hip_atomic_store(cnt, 0, __ATOMIC_RELAXED, __HIP_MEMORY_SCOPE_AGENT);
      __hip_atomic_fetch_add(gen, 1, __ATOMIC_ACQ_REL, __HIP_MEMORY_SCOPE_AGENT);
    } else {
      while (__hip_atomic_load(gen, __ATOMIC_ACQUIRE, __HIP_MEMORY_SCOPE_AGENT) == g)
        __builtin_amdgcn_s_sleep(1);
    }
  }
  __syncthreads();
}

// ---------------------------------------------------------------- recurrence
// Persistent kernel: 64 blocks (2 batch-tiles x 32 column-chunks) x 64 thr.
// Each block async-copies its 32-row slice of Whz/Whr/Whg (192 KB) into LDS
// once, then runs all 512 steps reading weight fragments from LDS.
// Step: phase1 z,r over full h(t-1); write r*h; sync;
//       phase2 g over full r*h; h_new = z*h+(1-z)*g; sync.
__global__ void gru_rec_kernel(const _Float16* __restrict__ gz,
                               const _Float16* __restrict__ gr,
                               const _Float16* __restrict__ gg,
                               const _Float16* __restrict__ Whz,
                               const _Float16* __restrict__ Whr,
                               const _Float16* __restrict__ Whg,
                               const float* __restrict__ bz,
                               const float* __restrict__ br,
                               const float* __restrict__ bg,
                               const float* __restrict__ h0,
                               int layer,
                               _Float16* __restrict__ h_cur,
                               _Float16* __restrict__ rh,
                               _Float16* __restrict__ seq_out,
                               float* __restrict__ hT,
                               int* sync_cnt, int* sync_gen) {
  extern __shared__ _Float16 smem[];     // [3][32 rows][H_] f16 = 192 KB
  const int nblocks = gridDim.x;         // 64
  const int bt = blockIdx.x >> 5;        // batch tile (0..1)
  const int nc = blockIdx.x & 31;        // 32-column chunk
  const int lane = threadIdx.x & 31;
  const int waveId = threadIdx.x >> 5;   // 0..1
  const int n_base = nc * 32;
  const int n0 = n_base + waveId * 16;

  // ---- async fill: 3 weight slices (32 rows x 1024) -> LDS
  {
    const _Float16* wsrc[3] = {Whz, Whr, Whg};
    for (int mat = 0; mat < 3; ++mat) {
      const _Float16* src = wsrc[mat] + (size_t)n_base * H_;
      _Float16* dst = smem + (size_t)mat * 32 * H_;
      for (int i = threadIdx.x; i < 32 * H_ / 8; i += blockDim.x) {
        unsigned lds_addr = (unsigned)(uintptr_t)(dst + i * 8);
        unsigned long long ga = (unsigned long long)(uintptr_t)(src + i * 8);
        asm volatile("global_load_async_to_lds_b128 %0, %1, off"
                     :: "v"(lds_addr), "v"(ga) : "memory");
      }
    }
    asm volatile("s_wait_asynccnt 0x0" ::: "memory");
    __syncthreads();
  }

  // ---- load initial hidden state for this block's (16 x 32) patch
  for (int i = threadIdx.x; i < 16 * 32; i += blockDim.x) {
    int m = bt * 16 + (i >> 5);
    int n = n_base + (i & 31);
    h_cur[(size_t)m * H_ + n] = (_Float16)h0[((size_t)m * L_ + layer) * H_ + n];
  }
  grid_sync(sync_cnt, sync_gen, nblocks);

  const int koff = (lane >> 4) << 3;
  const int half = lane >> 4;
  const int rowA = bt * 16 + (lane & 15);
  const int lrow = waveId * 16 + (lane & 15);     // row within LDS slice
  const _Float16* ap  = h_cur + (size_t)rowA * H_ + koff;
  const _Float16* rp  = rh    + (size_t)rowA * H_ + koff;
  const _Float16* lzp = smem + (size_t)0 * 32 * H_ + (size_t)lrow * H_ + koff;
  const _Float16* lrp = smem + (size_t)1 * 32 * H_ + (size_t)lrow * H_ + koff;
  const _Float16* lgp = smem + (size_t)2 * 32 * H_ + (size_t)lrow * H_ + koff;
  const int n = n0 + (lane & 15);
  const float bzv = bz[n], brv = br[n], bgv = bg[n];

  for (int t = 0; t < S_; ++t) {
    // ---- phase 1: z and r gates over full h(t-1); weights from LDS
    v8f az = {}, ar = {};
    for (int k = 0; k < H_; k += 32) {
      v16h a = ldfrag(ap + k);
      az = wmma_f16(a, ldfrag(lzp + k), az);
      ar = wmma_f16(a, ldfrag(lrp + k), ar);
    }
    float zv[8], hv[8];
#pragma unroll
    for (int r = 0; r < 8; ++r) {
      const int m = bt * 16 + r + 8 * half;
      const size_t gidx = ((size_t)m * S_ + t) * H_ + n;
      const float hval = (float)h_cur[(size_t)m * H_ + n];
      const float z  = 1.0f / (1.0f + __expf(-((float)gz[gidx] + az[r] + bzv)));
      const float rg = 1.0f / (1.0f + __expf(-((float)gr[gidx] + ar[r] + brv)));
      rh[(size_t)m * H_ + n] = (_Float16)(rg * hval);
      zv[r] = z; hv[r] = hval;
    }
    grid_sync(sync_cnt, sync_gen, nblocks);  // rh complete device-wide

    // ---- phase 2: candidate gate over full r*h, then h update
    v8f ag = {};
    for (int k = 0; k < H_; k += 32) {
      v16h a = ldfrag(rp + k);
      ag = wmma_f16(a, ldfrag(lgp + k), ag);
    }
#pragma unroll
    for (int r = 0; r < 8; ++r) {
      const int m = bt * 16 + r + 8 * half;
      const size_t gidx = ((size_t)m * S_ + t) * H_ + n;
      const float g  = tanhf((float)gg[gidx] + ag[r] + bgv);
      const float hn = zv[r] * hv[r] + (1.0f - zv[r]) * g;
      seq_out[gidx] = (_Float16)hn;
      h_cur[(size_t)m * H_ + n] = (_Float16)hn;
      if (t == S_ - 1) hT[((size_t)m * L_ + layer) * H_ + n] = hn;
    }
    grid_sync(sync_cnt, sync_gen, nblocks);  // h_cur complete before next step
  }
}

// ---------------------------------------------------------------- host side
static void cvt(const void* s, _Float16* d, int n, hipStream_t st) {
  cvt_f32_f16_kernel<<<dim3((n + 255) / 256), dim3(256), 0, st>>>(
      (const float*)s, d, n);
}

static void gemm(const _Float16* A, const _Float16* W, const float* bias,
                 _Float16* oh, float* of, int M, int N, int K, hipStream_t st) {
  int waves = (M / 64) * (N / 32);
  int blocks = (waves + 7) / 8;  // 8 waves (256 threads) per block
  wmma_gemm_kernel<<<dim3(blocks), dim3(256), 0, st>>>(A, W, bias, oh, of, M, N, K);
}

extern "C" void kernel_launch(void* const* d_in, const int* in_sizes, int n_in,
                              void* d_out, int out_size, void* d_ws, size_t ws_size,
                              hipStream_t stream) {
  (void)in_sizes; (void)n_in; (void)out_size; (void)ws_size;
  const float* x  = (const float*)d_in[0];
  const float* h0 = (const float*)d_in[1];
  const float* Wxz[2] = {(const float*)d_in[2],  (const float*)d_in[11]};
  const float* Whz[2] = {(const float*)d_in[3],  (const float*)d_in[12]};
  const float* bz [2] = {(const float*)d_in[4],  (const float*)d_in[13]};
  const float* Wxr[2] = {(const float*)d_in[5],  (const float*)d_in[14]};
  const float* Whr[2] = {(const float*)d_in[6],  (const float*)d_in[15]};
  const float* br [2] = {(const float*)d_in[7],  (const float*)d_in[16]};
  const float* Wxg[2] = {(const float*)d_in[8],  (const float*)d_in[17]};
  const float* Whg[2] = {(const float*)d_in[9],  (const float*)d_in[18]};
  const float* bg [2] = {(const float*)d_in[10], (const float*)d_in[19]};
  const float* Wy = (const float*)d_in[20];
  const float* by = (const float*)d_in[21];

  char* base = (char*)d_ws;
  size_t off = 0;
  auto carve = [&](size_t elems) -> _Float16* {
    _Float16* p = (_Float16*)(base + off);
    off += ((elems * sizeof(_Float16)) + 255) & ~(size_t)255;
    return p;
  };

  _Float16* xa  = carve((size_t)B_ * S_ * H_);   // layer input  (ping)
  _Float16* xb  = carve((size_t)B_ * S_ * H_);   // layer output (pong)
  _Float16* gzb = carve((size_t)B_ * S_ * H_);   // gate pre-activations
  _Float16* grb = carve((size_t)B_ * S_ * H_);
  _Float16* ggb = carve((size_t)B_ * S_ * H_);
  _Float16* wxz16[2], *whz16[2], *wxr16[2], *whr16[2], *wxg16[2], *whg16[2];
  const int ind[2] = {I_, H_};
  for (int l = 0; l < 2; ++l) {
    wxz16[l] = carve((size_t)H_ * ind[l]);
    whz16[l] = carve((size_t)H_ * H_);
    wxr16[l] = carve((size_t)H_ * ind[l]);
    whr16[l] = carve((size_t)H_ * H_);
    wxg16[l] = carve((size_t)H_ * ind[l]);
    whg16[l] = carve((size_t)H_ * H_);
  }
  _Float16* wy16 = carve((size_t)O_ * H_);
  _Float16* hcur = carve((size_t)B_ * H_);
  _Float16* rhb  = carve((size_t)B_ * H_);
  int* sync = (int*)(base + off); off += 256;

  // f32 -> f16 conversions (weights keep reference row-major (N,K) layout)
  cvt(x, xa, B_ * S_ * I_, stream);
  for (int l = 0; l < 2; ++l) {
    cvt(Wxz[l], wxz16[l], H_ * ind[l], stream);
    cvt(Whz[l], whz16[l], H_ * H_,     stream);
    cvt(Wxr[l], wxr16[l], H_ * ind[l], stream);
    cvt(Whr[l], whr16[l], H_ * H_,     stream);
    cvt(Wxg[l], wxg16[l], H_ * ind[l], stream);
    cvt(Whg[l], whg16[l], H_ * H_,     stream);
  }
  cvt(Wy, wy16, O_ * H_, stream);

  float* y_out  = (float*)d_out;
  float* hT_out = (float*)d_out + YSIZE;

  _Float16* cur = xa;
  _Float16* nxt = xb;
  const size_t rec_lds = (size_t)3 * 32 * H_ * sizeof(_Float16);  // 192 KB
  for (int l = 0; l < 2; ++l) {
    const int K = ind[l];
    // input projections as one big WMMA GEMM per gate (M = B*S)
    gemm(cur, wxz16[l], nullptr, gzb, nullptr, B_ * S_, H_, K, stream);
    gemm(cur, wxr16[l], nullptr, grb, nullptr, B_ * S_, H_, K, stream);
    gemm(cur, wxg16[l], nullptr, ggb, nullptr, B_ * S_, H_, K, stream);
    // persistent recurrence over S_ steps
    init_sync_kernel<<<dim3(1), dim3(1), 0, stream>>>(sync);
    gru_rec_kernel<<<dim3(64), dim3(64), rec_lds, stream>>>(
        gzb, grb, ggb, whz16[l], whr16[l], whg16[l],
        bz[l], br[l], bg[l], h0, l, hcur, rhb, nxt, hT_out,
        sync, sync + 1);
    _Float16* t = cur; cur = nxt; nxt = t;
  }
  // output projection: y = seq @ Wy.T + by  (f32 straight into d_out)
  gemm(cur, wy16, by, nullptr, y_out, B_ * S_, O_, H_, stream);
}
